// NT_Xent_Loss_17051020165451
// MI455X (gfx1250) — matrix operand order repaired and checked by reading, not compiled, not run on hardware
//
#include <hip/hip_runtime.h>
#include <hip/hip_bf16.h>

typedef __attribute__((ext_vector_type(16))) _Float16 v16h;
typedef __attribute__((ext_vector_type(8)))  _Float16 v8h;
typedef __attribute__((ext_vector_type(8)))  float    v8f;

#define DDIM    256
#define BROWS   4096
#define NROWS   8192
#define TINV    10.0f                   // 1 / temperature
#define L2E10   14.42695040888963f      // 10 * log2(e)

#define NC      64                      // B-panel columns per step
#define PSTR    264                     // padded LDS column stride (halfs): 132 dw = 4 mod 64 banks
#define CSPLIT  8                       // column splits across gridDim.y
#define CPB     (NROWS / CSPLIT)        // 1024 columns per block
#define NSTEPS  (CPB / NC)              // 16 panel steps
#define RPB     256                     // rows per block (8 waves x 2 tiles x 16)

// ---------------------------------------------------------------------------
// Kernel 1: row L2-normalize -> f16 matrix; zero the rowsum accumulator.
// ---------------------------------------------------------------------------
__global__ __launch_bounds__(256) void ntxent_normalize(const float* __restrict__ zi,
                                                        const float* __restrict__ zj,
                                                        _Float16* __restrict__ znh,
                                                        float* __restrict__ rowsum) {
    const int g = blockIdx.x * 256 + threadIdx.x;
    if (g < NROWS) rowsum[g] = 0.0f;

    const int lane = threadIdx.x & 31;
    const int wave = threadIdx.x >> 5;
    const int row  = blockIdx.x * 8 + wave;
    const float* src = (row < BROWS) ? (zi + (size_t)row * DDIM)
                                     : (zj + (size_t)(row - BROWS) * DDIM);
    float4 v0 = *(const float4*)(src + lane * 8);
    float4 v1 = *(const float4*)(src + lane * 8 + 4);
    float ss = v0.x * v0.x + v0.y * v0.y + v0.z * v0.z + v0.w * v0.w
             + v1.x * v1.x + v1.y * v1.y + v1.z * v1.z + v1.w * v1.w;
    #pragma unroll
    for (int off = 1; off < 32; off <<= 1) ss += __shfl_xor(ss, off, 32);
    const float inv = 1.0f / fmaxf(sqrtf(ss), 1e-8f);
    v8h o;
    o[0] = (_Float16)(v0.x * inv); o[1] = (_Float16)(v0.y * inv);
    o[2] = (_Float16)(v0.z * inv); o[3] = (_Float16)(v0.w * inv);
    o[4] = (_Float16)(v1.x * inv); o[5] = (_Float16)(v1.y * inv);
    o[6] = (_Float16)(v1.z * inv); o[7] = (_Float16)(v1.w * inv);
    *(v8h*)(znh + (size_t)row * DDIM + lane * 8) = o;
}

// ---------------------------------------------------------------------------
// Stage one 64-col x 256-K f16 panel (32KB) global -> LDS; 8 b128 per thread.
// ---------------------------------------------------------------------------
__device__ __forceinline__ void stage_panel(_Float16* __restrict__ dst,
                                            const _Float16* __restrict__ znh,
                                            int cb, int tid) {
    #pragma unroll
    for (int i = 0; i < 8; ++i) {
        const int q   = tid + i * 256;        // 0..2047 chunk id
        const int col = q >> 5;               // 0..63
        const int off = (q & 31) * 8;         // half offset 0..248
        v8h d = *(const v8h*)(znh + (size_t)(cb + col) * DDIM + off);
        *(v8h*)(dst + col * PSTR + off) = d;
    }
}

// ---------------------------------------------------------------------------
// Kernel 2: PURE fused sim-GEMM + exp-sum (no masks: diagonal & positive
// terms handled analytically in kernel 3). Block = 8 waves x 2 A-tiles =
// 256 rows; 64-col B panel shared via LDS, double-buffered; each B fragment
// feeds two WMMA chains. exp arg in [-2*L2E10, 0] => raw v_exp_f32 is safe.
// ---------------------------------------------------------------------------
__global__ __launch_bounds__(256) void ntxent_simloss(const _Float16* __restrict__ znh,
                                                      float* __restrict__ rowsum) {
    __shared__ __align__(16) _Float16 panel[2][NC * PSTR];
    const int tid  = threadIdx.x;
    const int lane = tid & 31;
    const int wave = tid >> 5;
    const int hi   = lane >> 4;
    const int l15  = lane & 15;
    const int m0   = blockIdx.x * RPB;
    const int cb0  = blockIdx.y * CPB;
    const int mt0  = m0 + wave * 32;      // A-tile 0 row base
    const int mt1  = mt0 + 16;            // A-tile 1 row base

    // --- Preload A fragments (ISA 16-bit A layout: lane L<16: M=L,
    // K={0..7,16..23}; lanes 16-31: M=L-16, K={8..15,24..31}).
    v16h a0[8], a1[8];
    {
        const _Float16* ar0 = znh + (size_t)(mt0 + l15) * DDIM;
        const _Float16* ar1 = znh + (size_t)(mt1 + l15) * DDIM;
        #pragma unroll
        for (int kc = 0; kc < 8; ++kc) {
            union { v16h v; v8h h[2]; } u;
            const _Float16* p0 = ar0 + kc * 32 + hi * 8;
            u.h[0] = *(const v8h*)(p0);
            u.h[1] = *(const v8h*)(p0 + 16);
            a0[kc] = u.v;
            const _Float16* p1 = ar1 + kc * 32 + hi * 8;
            u.h[0] = *(const v8h*)(p1);
            u.h[1] = *(const v8h*)(p1 + 16);
            a1[kc] = u.v;
        }
    }

    v8f esum0 = {0.f,0.f,0.f,0.f,0.f,0.f,0.f,0.f};
    v8f esum1 = {0.f,0.f,0.f,0.f,0.f,0.f,0.f,0.f};

    stage_panel(&panel[0][0], znh, cb0, tid);
    __syncthreads();

    #pragma unroll 1
    for (int st = 0; st < NSTEPS; ++st) {
        const int buf = st & 1;
        if (st + 1 < NSTEPS)
            stage_panel(&panel[buf ^ 1][0], znh, cb0 + (st + 1) * NC, tid);
        if (st + 2 < NSTEPS)
            __builtin_prefetch(znh + (size_t)(cb0 + (st + 2) * NC) * DDIM, 0, 3);

        const _Float16* pp = &panel[buf][0];
        #pragma unroll
        for (int ct = 0; ct < 4; ++ct) {
            const _Float16* lp = pp + (ct * 16 + l15) * PSTR + hi * 8;

            // Software-pipelined B chunks: load kc+1 before consuming kc.
            union { v16h v; v8h h[2]; } ub;
            ub.h[0] = *(const v8h*)(lp);
            ub.h[1] = *(const v8h*)(lp + 16);
            v16h bcur = ub.v;

            v8f c0 = {0.f,0.f,0.f,0.f,0.f,0.f,0.f,0.f};
            v8f c1 = {0.f,0.f,0.f,0.f,0.f,0.f,0.f,0.f};
            #pragma unroll
            for (int kc = 0; kc < 8; ++kc) {
                v16h bnext;
                if (kc + 1 < 8) {
                    union { v16h v; v8h h[2]; } un;
                    un.h[0] = *(const v8h*)(lp + (kc + 1) * 32);
                    un.h[1] = *(const v8h*)(lp + (kc + 1) * 32 + 16);
                    bnext = un.v;
                }
                c0 = __builtin_amdgcn_wmma_f32_16x16x32_f16(
                        false, a0[kc], false, bcur, (short)0, c0, false, false);
                c1 = __builtin_amdgcn_wmma_f32_16x16x32_f16(
                        false, a1[kc], false, bcur, (short)0, c1, false, false);
                if (kc + 1 < 8) bcur = bnext;
            }
            // Branch-free epilogue. Separate e-vectors keep the 16 v_exp_f32
            // chains independent (no serial temp-register dependency).
            v8f e0, e1;
            #pragma unroll
            for (int r = 0; r < 8; ++r) {
                e0[r] = __builtin_amdgcn_exp2f(fmaf(c0[r], L2E10, -L2E10));
                e1[r] = __builtin_amdgcn_exp2f(fmaf(c1[r], L2E10, -L2E10));
            }
            esum0 += e0;
            esum1 += e1;
        }
        __syncthreads();
    }

    // Cross-lane reduce (rows are wave-exclusive) + global partial accumulate.
    #pragma unroll
    for (int r = 0; r < 8; ++r) {
        float v0 = esum0[r];
        float v1 = esum1[r];
        #pragma unroll
        for (int off = 1; off < 16; off <<= 1) {
            v0 += __shfl_xor(v0, off, 32);
            v1 += __shfl_xor(v1, off, 32);
        }
        if (l15 == 0) {
            atomicAdd(&rowsum[mt0 + hi * 8 + r], v0);   // global_atomic_add_f32
            atomicAdd(&rowsum[mt1 + hi * 8 + r], v1);
        }
    }
}

// ---------------------------------------------------------------------------
// Kernel 3: per-row loss term. One wave per row:
//   d_i   = |zn_i|^2            (diagonal correction)
//   pos_i = (1/T) zn_i . zn_{(i+B)%N}
//   loss_i = log(rowsum_i - exp(d_i/T - 10)) + 10 - pos_i
// ---------------------------------------------------------------------------
__global__ __launch_bounds__(256) void ntxent_rowterm(const _Float16* __restrict__ znh,
                                                      const float* __restrict__ rowsum,
                                                      float* __restrict__ loss) {
    const int lane = threadIdx.x & 31;
    const int wave = threadIdx.x >> 5;
    const int i = blockIdx.x * 8 + wave;
    const int j = (i + BROWS) & (NROWS - 1);
    v8h av = *(const v8h*)(znh + (size_t)i * DDIM + lane * 8);
    v8h bv = *(const v8h*)(znh + (size_t)j * DDIM + lane * 8);
    float d = 0.0f, p = 0.0f;
    #pragma unroll
    for (int e = 0; e < 8; ++e) {
        const float fa = (float)av[e];
        const float fb = (float)bv[e];
        d = fmaf(fa, fa, d);
        p = fmaf(fa, fb, p);
    }
    #pragma unroll
    for (int off = 1; off < 32; off <<= 1) {
        d += __shfl_xor(d, off, 32);
        p += __shfl_xor(p, off, 32);
    }
    if (lane == 0) {
        const float denom = rowsum[i] - __builtin_amdgcn_exp2f(fmaf(d, L2E10, -L2E10));
        loss[i] = (__logf(denom) + TINV) - TINV * p;
    }
}

// ---------------------------------------------------------------------------
// Kernel 4: loss = mean(loss_i)
// ---------------------------------------------------------------------------
__global__ __launch_bounds__(256) void ntxent_mean(const float* __restrict__ loss,
                                                   float* __restrict__ out) {
    __shared__ float sbuf[256];
    float acc = 0.0f;
    for (int i = threadIdx.x; i < NROWS; i += 256) acc += loss[i];
    sbuf[threadIdx.x] = acc;
    __syncthreads();
    #pragma unroll
    for (int s = 128; s > 0; s >>= 1) {
        if (threadIdx.x < s) sbuf[threadIdx.x] += sbuf[threadIdx.x + s];
        __syncthreads();
    }
    if (threadIdx.x == 0) out[0] = sbuf[0] / (float)NROWS;
}

extern "C" void kernel_launch(void* const* d_in, const int* in_sizes, int n_in,
                              void* d_out, int out_size, void* d_ws, size_t ws_size,
                              hipStream_t stream) {
    const float* zi = (const float*)d_in[0];
    const float* zj = (const float*)d_in[1];
    float* out = (float*)d_out;

    // Workspace: [znh f16 8192x256 = 4MB][rowsum f32 8192][loss f32 8192]
    _Float16* znh   = (_Float16*)d_ws;
    float*   rowsum = (float*)((char*)d_ws + (size_t)NROWS * DDIM * sizeof(_Float16));
    float*   loss   = rowsum + NROWS;

    ntxent_normalize<<<NROWS / 8, 256, 0, stream>>>(zi, zj, znh, rowsum);
    ntxent_simloss<<<dim3(NROWS / RPB, CSPLIT), 256, 0, stream>>>(znh, rowsum);
    ntxent_rowterm<<<NROWS / 8, 256, 0, stream>>>(znh, rowsum, loss);
    ntxent_mean<<<1, 256, 0, stream>>>(loss, out);
}
